// SNNLocalizer_25159918420194
// MI455X (gfx1250) — matrix operand order repaired and checked by reading, not compiled, not run on hardware
//
#include <hip/hip_runtime.h>

// ---------------------------------------------------------------------------
// SNN localizer forward, MI455X (gfx1250, wave32, WMMA).
//
// T=500 sequential LIF steps; recurrence forces per-step serialization, so the
// design goal is the shortest per-step critical path:
//   - W1 B-fragments register-resident for all 500 steps (zero weight traffic)
//   - spike tile for step t+1 held in VGPRs; split-barrier in inline asm waits
//     only DScnt, so those global loads stay in flight across the barrier
//     (t+2 prefetched toward L2)
//   - sched_group_barrier forces a DS-read/WMMA interleave (3 A-tiles in
//     flight) so ds_load latency hides behind the WMMA accumulation chain
//   - W2 fragments live in LDS (cold path, keeps VGPR pressure low)
// ---------------------------------------------------------------------------

typedef __attribute__((ext_vector_type(16))) _Float16 v16h;
typedef __attribute__((ext_vector_type(8)))  _Float16 v8h;
typedef __attribute__((ext_vector_type(8)))  float    v8f;

#define T_STEPS 500
#define B_TOT   256
#define F_IN    512
#define H1_DIM  128
#define H2_DIM  64
#define BT      16          // batch tile per workgroup
#define XPAD    520         // 512 + 8 halves row pad (16B aligned, bank-spread)
#define SPAD    136         // 128 + 8 halves row pad

__device__ __constant__ float c_beta1 = 0.8187307530779818f;  // exp(-1/5)
__device__ __constant__ float c_beta2 = 0.9512294245007140f;  // exp(-1/20)
#define THR1 1.2f
#define THR2 1.0f

// Split workgroup barrier that only drains DScnt (LDS visibility). Unlike
// __syncthreads(), this does NOT wait LOADcnt, so in-flight global loads
// (next-step spike tile -> private VGPRs) survive across the barrier.
__device__ __forceinline__ void wg_barrier_lds() {
    asm volatile("s_wait_dscnt 0x0\n\t"
                 "s_barrier_signal -1\n\t"
                 "s_barrier_wait -1" ::: "memory");
}

// Build a 16x32 f16 A-fragment from a padded row-major f16 LDS tile.
// Lane layout: row M = lane%16; elems 0..15 -> K = kb + half*16 + i
// (two 16B ds_load_b128).
__device__ __forceinline__ v16h load_afrag(const _Float16* base, int rowstride,
                                           int l16, int half, int kb) {
    const _Float16* p = base + l16 * rowstride + kb + half * 16;
    v8h lo = *(const v8h*)(p);
    v8h hi = *(const v8h*)(p + 8);
    v16h a;
    #pragma unroll
    for (int i = 0; i < 8; ++i) { a[i] = lo[i]; a[i + 8] = hi[i]; }
    return a;
}

__global__ __launch_bounds__(256, 1)
void SNNLocalizer_fwd_kernel(const float* __restrict__ spikes,
                             const float* __restrict__ W1, const float* __restrict__ b1,
                             const float* __restrict__ W2, const float* __restrict__ b2,
                             const float* __restrict__ Wr, const float* __restrict__ br,
                             float* __restrict__ out, float* __restrict__ partial) {
    __shared__ __align__(16) _Float16 xbuf[BT * XPAD];    // f16 staged x_t tile
    __shared__ __align__(16) _Float16 s1buf[BT * SPAD];   // f16 spk1 tile
    __shared__ __align__(16) _Float16 w2f[4 * 4 * 32 * 16]; // W2 B-frags, 16 KB
    __shared__ __align__(16) float    m2s[BT * H2_DIM];   // mem2_sum staging
    __shared__ float red[256];                            // scalar reduction

    const int tid  = threadIdx.x;
    const int wave = tid >> 5;        // 0..7
    const int lane = tid & 31;        // wave32 lane
    const int l16  = lane & 15;
    const int half = lane >> 4;       // 0: lanes 0-15, 1: lanes 16-31
    const int bt0  = blockIdx.x * BT;

    // ---- W1 B-fragments (16x16x32 f16 layout), register-resident ----
    // B[32x16]: lane holds column N = lane%16; f16 elem i holds
    // K = kt*32 + half*16 + i  -> 16 consecutive f32 from W1 row N.
    v16h bw1[16];
    {
        const int n = wave * 16 + l16;                // H1 column
        #pragma unroll
        for (int kt = 0; kt < 16; ++kt) {
            const float* src = W1 + n * F_IN + kt * 32 + half * 16;
            v16h b;
            #pragma unroll
            for (int i = 0; i < 16; ++i) b[i] = (_Float16)src[i];
            bw1[kt] = b;
        }
    }
    // ---- W2 B-fragments -> LDS (cold path; keeps VGPR pressure low) ----
    if (wave < 4) {
        const int n2 = wave * 16 + l16;               // H2 column
        #pragma unroll
        for (int kt = 0; kt < 4; ++kt) {
            const float* src = W2 + n2 * H1_DIM + kt * 32 + half * 16;
            v16h b;
            #pragma unroll
            for (int i = 0; i < 16; ++i) b[i] = (_Float16)src[i];
            _Float16* dst = w2f + ((wave * 4 + kt) * 32 + lane) * 16;
            *(v8h*)(dst) = *(v8h*)&b;
            *(v8h*)(dst + 8) = *((v8h*)&b + 1);
        }
    }

    const float bias1 = b1[wave * 16 + l16];
    const float bias2 = b2[(wave & 3) * 16 + l16];

    // Recurrent state in WMMA C layout: f32 elem v <-> row M = v + 8*half,
    // col N = tile*16 + l16. Lives in VGPRs across all 500 steps.
    v8f mem1 = {}; v8f mem2 = {}; v8f m2sum = {};
    float cnt1 = 0.0f, cnt2 = 0.0f;

    const int xrow = tid >> 4;          // 0..15 (batch row within tile)
    const int xcol = (tid & 15) * 32;   // 32-float column segment
    const size_t tile_off = (size_t)(bt0 + xrow) * F_IN + xcol;
    const size_t step_stride = (size_t)B_TOT * F_IN;

    // prologue: global loads for step 0 (f32, held in VGPRs)
    float4 xr[8];
    {
        const float4* src = (const float4*)(spikes + tile_off);
        #pragma unroll
        for (int i = 0; i < 8; ++i) xr[i] = src[i];
    }

    for (int t = 0; t < T_STEPS; ++t) {
        // ---- stage: cvt step-t registers -> f16 LDS tile ----
        {
            _Float16* dst = xbuf + xrow * XPAD + xcol;
            #pragma unroll
            for (int i = 0; i < 8; ++i) {
                float4 v = xr[i];
                dst[4 * i + 0] = (_Float16)v.x;
                dst[4 * i + 1] = (_Float16)v.y;
                dst[4 * i + 2] = (_Float16)v.z;
                dst[4 * i + 3] = (_Float16)v.w;
            }
        }
        // ---- issue step t+1 global loads; they stay in flight across the
        //      barrier (only DScnt drains) and are waited at next cvt ----
        if (t + 1 < T_STEPS) {
            const float4* src =
                (const float4*)(spikes + tile_off + (size_t)(t + 1) * step_stride);
            #pragma unroll
            for (int i = 0; i < 8; ++i) xr[i] = src[i];
            if (t + 2 < T_STEPS)  // warm L2 two steps ahead
                __builtin_prefetch(spikes + tile_off + (size_t)(t + 2) * step_stride, 0, 1);
        }
        wg_barrier_lds();   // barrier A: xbuf(t) visible

        // ---- layer 1: 16 WMMAs over K=512 ----
        v8f acc = {};
        #pragma unroll
        for (int kt = 0; kt < 16; ++kt) {
            v16h a = load_afrag(xbuf, XPAD, l16, half, kt * 32);
            acc = __builtin_amdgcn_wmma_f32_16x16x32_f16(
                false, a, false, bw1[kt], (short)0, acc, false, false);
        }
#if __has_builtin(__builtin_amdgcn_sched_group_barrier)
        // Pipeline directive: keep 3 A-tiles (6 ds_reads) in flight ahead of
        // each WMMA so s_wait_dscnt becomes partial and LDS latency hides
        // behind the WMMA accumulation chain.
        //   DS_READ x6, [WMMA x1, DS_READ x2] x13, WMMA x3
        __builtin_amdgcn_sched_group_barrier(0x100, 6, 0);   // 6 DS reads
        #pragma unroll
        for (int i = 0; i < 13; ++i) {
            __builtin_amdgcn_sched_group_barrier(0x008, 1, 0);  // 1 WMMA
            __builtin_amdgcn_sched_group_barrier(0x100, 2, 0);  // 2 DS reads
        }
        __builtin_amdgcn_sched_group_barrier(0x008, 3, 0);   // tail WMMAs
#endif

        // LIF layer 1 + stage spk1 (f16, row-major padded)
        {
            _Float16* sb = s1buf + wave * 16 + l16;
            #pragma unroll
            for (int v = 0; v < 8; ++v) {
                float m  = c_beta1 * mem1[v] + acc[v] + bias1;
                float sp = (m > THR1) ? 1.0f : 0.0f;   // spike(m - THR1)
                mem1[v]  = m - sp * THR1;              // reset by subtraction
                cnt1    += sp;
                sb[(v + half * 8) * SPAD] = (_Float16)sp;
            }
        }
        wg_barrier_lds();   // barrier B: spk1 visible; layer-1 xbuf reads done

        // ---- layer 2 on waves 0..3: 4 WMMAs over K=128 ----
        if (wave < 4) {
            v8f acc2 = {};
            #pragma unroll
            for (int kt = 0; kt < 4; ++kt) {
                v16h a = load_afrag(s1buf, SPAD, l16, half, kt * 32);
                const _Float16* wp = w2f + ((wave * 4 + kt) * 32 + lane) * 16;
                v16h b;
                *(v8h*)&b = *(const v8h*)(wp);
                *((v8h*)&b + 1) = *(const v8h*)(wp + 8);
                acc2 = __builtin_amdgcn_wmma_f32_16x16x32_f16(
                    false, a, false, b, (short)0, acc2, false, false);
            }
            #pragma unroll
            for (int v = 0; v < 8; ++v) {
                float m  = c_beta2 * mem2[v] + acc2[v] + bias2;
                float sp = (m > THR2) ? 1.0f : 0.0f;
                m        = m - sp * THR2;
                mem2[v]  = m;
                m2sum[v] += m;
                cnt2    += sp;
            }
        }
        // no barrier here: next-iteration writes to xbuf/s1buf only happen
        // after each wave passes barrier B(t)/A(t+1) respectively.
    }
    __syncthreads();

    // ---- readout: out[b,:] = (mem2_sum/T) @ Wr^T + br for this batch tile ----
    if (wave < 4) {
        #pragma unroll
        for (int v = 0; v < 8; ++v)
            m2s[(v + half * 8) * H2_DIM + wave * 16 + l16] = m2sum[v];
    }
    red[tid] = cnt1;
    __syncthreads();
    if (tid < 32) {
        const int b = tid >> 1, o = tid & 1;
        const float* wr = Wr + o * H2_DIM;
        float s = 0.0f;
        #pragma unroll
        for (int h = 0; h < H2_DIM; ++h) s += m2s[b * H2_DIM + h] * wr[h];
        out[(bt0 + b) * 2 + o] = s * (1.0f / (float)T_STEPS) + br[o];
    }
    if (tid == 0) {
        float s = 0.0f;
        for (int i = 0; i < 256; ++i) s += red[i];
        partial[blockIdx.x] = s;          // spk1 count for this batch tile
    }
    __syncthreads();
    red[tid] = cnt2;                      // zero on waves 4..7
    __syncthreads();
    if (tid == 0) {
        float s = 0.0f;
        for (int i = 0; i < 256; ++i) s += red[i];
        partial[16 + blockIdx.x] = s;     // spk2 count
    }
}

// Deterministic cross-workgroup reduction of spike-rate scalars.
__global__ void SNNLocalizer_finalize_kernel(const float* __restrict__ partial,
                                             float* __restrict__ out) {
    if (threadIdx.x == 0) {
        float s1 = 0.0f, s2 = 0.0f;
        for (int i = 0; i < 16; ++i) { s1 += partial[i]; s2 += partial[16 + i]; }
        out[512] = s1 / ((float)T_STEPS * (float)B_TOT * (float)H1_DIM);
        out[513] = s2 / ((float)T_STEPS * (float)B_TOT * (float)H2_DIM);
    }
}

extern "C" void kernel_launch(void* const* d_in, const int* in_sizes, int n_in,
                              void* d_out, int out_size, void* d_ws, size_t ws_size,
                              hipStream_t stream) {
    const float* spikes = (const float*)d_in[0];
    const float* W1 = (const float*)d_in[1];
    const float* b1 = (const float*)d_in[2];
    const float* W2 = (const float*)d_in[3];
    const float* b2 = (const float*)d_in[4];
    const float* Wr = (const float*)d_in[5];
    const float* br = (const float*)d_in[6];
    float* out     = (float*)d_out;
    float* partial = (float*)d_ws;   // 32 floats of scratch

    SNNLocalizer_fwd_kernel<<<B_TOT / BT, 256, 0, stream>>>(
        spikes, W1, b1, W2, b2, Wr, br, out, partial);
    SNNLocalizer_finalize_kernel<<<1, 32, 0, stream>>>(partial, out);
}
